// SlidingWindowAttention_11544872091864
// MI455X (gfx1250) — matrix operand order repaired and checked by reading, not compiled, not run on hardware
//
#include <hip/hip_runtime.h>
#include <hip/hip_bf16.h>
#include <stdint.h>

// ---------------------------------------------------------------------------
// Sliding-window attention for MI455X (gfx1250, wave32, WMMA).
// Pipeline: RMSNorm->bf16 | w->bf16 | QKV GEMM (WMMA bf16, async LDS copies) |
//           RoPE+split | flash attention (windowed, WMMA bf16) | out GEMM.
// ---------------------------------------------------------------------------

typedef __bf16 bf16;
typedef __attribute__((ext_vector_type(16))) __bf16 v16bf;
typedef __attribute__((ext_vector_type(8)))  float  v8f;

#define N_SEQ    4096
#define D_MODEL  1024
#define HEADS    8
#define DIM_HEAD 64
#define DINNER   512       // HEADS * DIM_HEAD
#define QKV_N    1536      // 3 * DINNER
#define WINDOW   128
#define PERSIST  4
#define RMS_EPS  1.1920929e-07f

__device__ __forceinline__ bf16 f2bf(float f) { return (bf16)f; }

// ---------------------------------------------------------------------------
// CDNA5 async global->LDS copy (ASYNCcnt path, CDNA5 ISA).
// LDS dest operand is the wave-relative LDS byte offset == low 32 bits of the
// flat pointer (flat LDS aperture mapping truncates to addr[31:0]).
__device__ __forceinline__ void async_copy_b128_to_lds(const bf16* gsrc, bf16* ldst) {
  const unsigned int       lds_off = (unsigned int)(size_t)ldst;
  const unsigned long long gaddr   = (unsigned long long)(size_t)gsrc;
  asm volatile("global_load_async_to_lds_b128 %0, %1, off"
               :: "v"(lds_off), "v"(gaddr) : "memory");
}
__device__ __forceinline__ void wait_asynccnt0() {
  asm volatile("s_wait_asynccnt 0" ::: "memory");
}

// ---------------------------------------------------------------------------
// Fragment loaders (wave32), contiguous k order k = 16*(lane/16) + s, same for A and B.
// A (16x32, 16-bit): lane m = lane&15; k runs [kb,kb+16),
//   kb = (lane<16 ? 0 : 16).  Source row-major [m][k], 'stride' bf16 per row.
__device__ __forceinline__ v16bf load_a_frag(const bf16* base, int stride) {
  const int lane = threadIdx.x & 31;
  const int m  = lane & 15;
  const int kb = (lane & 16) ? 16 : 0;
  const uint32_t* p0 = (const uint32_t*)(base + m * stride + kb);
  const uint32_t* p1 = (const uint32_t*)(base + m * stride + kb + 8);
  union { v16bf v; uint32_t u[8]; } t;
#pragma unroll
  for (int i = 0; i < 4; i++) { t.u[i] = p0[i]; t.u[4 + i] = p1[i]; }
  return t.v;
}

// B (32x16, 16-bit): lane n = lane&15; k = kb + h, kb = (lane<16 ? 0 : 16).
// Source laid out [n][k] (contiguous k per n) -> 8 dword reads per lane.
__device__ __forceinline__ v16bf load_b_frag_nk(const bf16* base, int stride) {
  const int lane = threadIdx.x & 31;
  const int n  = lane & 15;
  const int kb = (lane & 16) ? 16 : 0;
  const uint32_t* p = (const uint32_t*)(base + n * stride + kb);
  union { v16bf v; uint32_t u[8]; } t;
#pragma unroll
  for (int i = 0; i < 8; i++) t.u[i] = p[i];
  return t.v;
}

// Same B fragment, source laid out [k][n] (row-major K) -> strided reads
// (coalesced across lanes: 16 consecutive bf16 per k-row).
__device__ __forceinline__ v16bf load_b_frag_kn(const bf16* base, int stride) {
  const int lane = threadIdx.x & 31;
  const int n  = lane & 15;
  const int kb = (lane & 16) ? 16 : 0;
  union { v16bf v; bf16 e[16]; } t;
#pragma unroll
  for (int h = 0; h < 16; h++) t.e[h] = base[(kb + h) * stride + n];
  return t.v;
}

// ---------------------------------------------------------------------------
// RMSNorm (fp32 in) -> bf16 out.  One block per row of 1024.
__global__ __launch_bounds__(256) void rmsnorm_kernel(
    const float* __restrict__ seq, const float* __restrict__ g,
    bf16* __restrict__ xb) {
  const int row = blockIdx.x;
  const float* x = seq + (size_t)row * D_MODEL;
  float ss = 0.f;
#pragma unroll
  for (int i = threadIdx.x; i < D_MODEL; i += 256) { float v = x[i]; ss += v * v; }
#pragma unroll
  for (int m = 16; m >= 1; m >>= 1) ss += __shfl_xor(ss, m, 32);
  __shared__ float red[8];
  if ((threadIdx.x & 31) == 0) red[threadIdx.x >> 5] = ss;
  __syncthreads();
  float tot = red[0];
#pragma unroll
  for (int i = 1; i < 8; i++) tot += red[i];
  const float inv = rsqrtf(tot * (1.0f / D_MODEL) + RMS_EPS);
  for (int i = threadIdx.x; i < D_MODEL; i += 256)
    xb[(size_t)row * D_MODEL + i] = f2bf(x[i] * inv * g[i]);
}

// fp32 -> bf16 (weights, one-time)
__global__ void cvt_bf16_kernel(const float* __restrict__ in,
                                bf16* __restrict__ out, int n) {
  int i = blockIdx.x * blockDim.x + threadIdx.x;
  if (i < n) out[i] = f2bf(in[i]);
}

// ---------------------------------------------------------------------------
// Tiled bf16 WMMA GEMM: C[M,N](f32) = A[M,K](bf16, row-major) @ B[K,N](bf16).
// Block = 256 thr (8 waves), tile 128x128, K-step 32.  Wave -> 32x64 subtile.
// A tile staged with GLOBAL_LOAD_ASYNC_TO_LDS_B128 (ASYNCcnt); B tile goes
// through registers because it is transposed into [n][k] on the way to LDS.
#define BM 128
#define BN 128
#define BK 32
#define TS 48   // LDS row stride (bf16): 32 payload + 16 pad, 96B rows

template <int LDA, int LDB, int LDC, int K>
__global__ __launch_bounds__(256) void gemm_bf16_kernel(
    const bf16* __restrict__ A, const bf16* __restrict__ B,
    float* __restrict__ C) {
  __shared__ bf16 As[BM * TS];      // [m][k]
  __shared__ bf16 Bt[BN * TS];      // [n][k] (transposed tile)
  const int tid   = threadIdx.x;
  const int mBase = blockIdx.y * BM;
  const int nBase = blockIdx.x * BN;
  const int wave  = tid >> 5, lane = tid & 31;
  const int wm = (wave & 3) * 32;
  const int wn = (wave >> 2) * 64;
  v8f acc[2][4] = {};

#pragma unroll 1
  for (int k0 = 0; k0 < K; k0 += BK) {
#pragma unroll
    for (int i = 0; i < 2; i++) {
      const int c = tid + i * 256;          // 512 16-byte chunks per tile
      {   // A: async DMA straight into LDS (no VGPR round-trip)
        const int row = c >> 2, coff = (c & 3) * 8;
        async_copy_b128_to_lds(A + (size_t)(mBase + row) * LDA + k0 + coff,
                               &As[row * TS + coff]);
      }
      {   // B: load row-major chunk, scatter transposed into Bt[n][k]
        const int krow = c >> 4, noff = (c & 15) * 8;
        const bf16* src = B + (size_t)(k0 + krow) * LDB + nBase + noff;
        union { uint4 q; bf16 e[8]; } v;
        v.q = *(const uint4*)src;
        if (k0 + BK < K) __builtin_prefetch(src + (size_t)BK * LDB, 0, 0);
#pragma unroll
        for (int j = 0; j < 8; j++) Bt[(noff + j) * TS + krow] = v.e[j];
      }
    }
    wait_asynccnt0();                       // A tile landed in LDS
    __syncthreads();

    v16bf af[2], bfr[4];
#pragma unroll
    for (int mt = 0; mt < 2; mt++)
      af[mt] = load_a_frag(&As[(wm + mt * 16) * TS], TS);
#pragma unroll
    for (int nt = 0; nt < 4; nt++)
      bfr[nt] = load_b_frag_nk(&Bt[(wn + nt * 16) * TS], TS);
#pragma unroll
    for (int mt = 0; mt < 2; mt++)
#pragma unroll
      for (int nt = 0; nt < 4; nt++)
        acc[mt][nt] = __builtin_amdgcn_wmma_f32_16x16x32_bf16(
            false, af[mt], false, bfr[nt], (short)0, acc[mt][nt], false, false);
    __syncthreads();
  }

  // C/D layout: lane<16 -> M = e, lane>=16 -> M = 8+e; N = lane&15.
  const int hi8 = (lane & 16) ? 8 : 0;
  const int nc  = lane & 15;
#pragma unroll
  for (int mt = 0; mt < 2; mt++)
#pragma unroll
    for (int nt = 0; nt < 4; nt++)
#pragma unroll
      for (int e = 0; e < 8; e++)
        C[(size_t)(mBase + wm + mt * 16 + hi8 + e) * LDC
          + nBase + wn + nt * 16 + nc] = acc[mt][nt][e];
}

// ---------------------------------------------------------------------------
// RoPE (interleaved pairs) + split qkv[N,1536](f32) -> q/k/v [H][N][Dh] bf16.
__global__ __launch_bounds__(256) void rope_split_kernel(
    const float* __restrict__ qkv,
    bf16* __restrict__ qb, bf16* __restrict__ kb, bf16* __restrict__ vb) {
  const int idx = blockIdx.x * 256 + threadIdx.x;    // N * 256 threads total
  const int n = idx >> 8;
  const int c = idx & 255;
  const int h = c >> 5;            // 32 pairs per head
  const int p = c & 31;
  const int d = p * 2;
  const float* row = qkv + (size_t)n * QKV_N;
  const float q0 = row[h * 64 + d],              q1 = row[h * 64 + d + 1];
  const float k0 = row[DINNER + h * 64 + d],     k1 = row[DINNER + h * 64 + d + 1];
  const float v0 = row[2 * DINNER + h * 64 + d], v1 = row[2 * DINNER + h * 64 + d + 1];
  const float freq = (float)n * __powf(10000.f, -(float)d * (1.0f / DIM_HEAD));
  float s, cs; __sincosf(freq, &s, &cs);
  const size_t o = ((size_t)h * N_SEQ + n) * DIM_HEAD + d;
  qb[o]     = f2bf(q0 * cs - q1 * s);
  qb[o + 1] = f2bf(q1 * cs + q0 * s);
  kb[o]     = f2bf(k0 * cs - k1 * s);
  kb[o + 1] = f2bf(k1 * cs + k0 * s);
  vb[o]     = f2bf(v0);
  vb[o + 1] = f2bf(v1);
}

// ---------------------------------------------------------------------------
// Flash attention, windowed.  Block = 128 thr (4 waves), 64 query rows per WG
// (16 per wave).  Key blocks of 32: j=0 (persist) then the sliding window.
#define PSTR 40   // LDS stride for P tile (bf16); 80B rows

__global__ __launch_bounds__(128) void attn_kernel(
    const bf16* __restrict__ qb, const bf16* __restrict__ kb,
    const bf16* __restrict__ vb, bf16* __restrict__ ob) {
  __shared__ bf16 Pb[4 * 16 * PSTR];
  const int qblk = blockIdx.x;
  const int h    = blockIdx.y;
  const int wave = threadIdx.x >> 5, lane = threadIdx.x & 31;
  const int q0    = qblk * 64;
  const int qrow0 = q0 + wave * 16;
  const int hi8 = (lane & 16) ? 8 : 0;
  const int nc  = lane & 15;
  const bf16* Qh = qb + (size_t)h * N_SEQ * DIM_HEAD;
  const bf16* Kh = kb + (size_t)h * N_SEQ * DIM_HEAD;
  const bf16* Vh = vb + (size_t)h * N_SEQ * DIM_HEAD;

  // Q fragments held in VGPRs for the whole key loop (Dh=64 -> two K-steps).
  const v16bf qf0 = load_a_frag(Qh + (size_t)qrow0 * DIM_HEAD, DIM_HEAD);
  const v16bf qf1 = load_a_frag(Qh + (size_t)qrow0 * DIM_HEAD + 32, DIM_HEAD);

  v8f O[4] = {};
  float mrow[8], lrow[8];
#pragma unroll
  for (int e = 0; e < 8; e++) { mrow[e] = -1e30f; lrow[e] = 0.f; }
  bf16* Pw = &Pb[wave * 16 * PSTR];
  const float scale = 0.125f;          // 64^-0.5
  const int qhi = q0 + 63;
  int jstart = q0 - WINDOW;
  if (jstart < 32) jstart = 32;
  jstart &= ~31;

  for (int it = -1;; ++it) {
    const int j = (it < 0) ? 0 : (jstart + it * 32);
    if (it >= 0 && j > qhi) break;

    // S[t] = Q(16xDh) @ K^T(Dh x 16), keys j+16t .. j+16t+15.
    // K is [key][dim] row-major == [n][k] layout for B fragments.
    v8f S[2] = {};
#pragma unroll
    for (int t = 0; t < 2; t++) {
      const bf16* kbase = Kh + (size_t)(j + t * 16) * DIM_HEAD;
      v16bf kf0 = load_b_frag_nk(kbase, DIM_HEAD);
      v16bf kf1 = load_b_frag_nk(kbase + 32, DIM_HEAD);
      S[t] = __builtin_amdgcn_wmma_f32_16x16x32_bf16(
          false, qf0, false, kf0, (short)0, S[t], false, false);
      S[t] = __builtin_amdgcn_wmma_f32_16x16x32_bf16(
          false, qf1, false, kf1, (short)0, S[t], false, false);
    }

    // scale + causal/persist/window mask
#pragma unroll
    for (int t = 0; t < 2; t++)
#pragma unroll
      for (int e = 0; e < 8; e++) {
        const int r    = qrow0 + hi8 + e;
        const int kcol = j + t * 16 + nc;
        const bool ok  = (kcol <= r) && ((kcol < PERSIST) || (r - kcol <= WINDOW));
        S[t][e] = ok ? S[t][e] * scale : -1e30f;
      }

    // online softmax (rows span 16-lane halves; xor masks 8..1 stay in-half)
#pragma unroll
    for (int e = 0; e < 8; e++) {
      float tm = fmaxf(S[0][e], S[1][e]);
#pragma unroll
      for (int m = 8; m >= 1; m >>= 1) tm = fmaxf(tm, __shfl_xor(tm, m, 32));
      const float mnew  = fmaxf(mrow[e], tm);
      const float alpha = __expf(mrow[e] - mnew);
      const float p0 = __expf(S[0][e] - mnew);
      const float p1 = __expf(S[1][e] - mnew);
      S[0][e] = p0; S[1][e] = p1;
      float rs = p0 + p1;
#pragma unroll
      for (int m = 8; m >= 1; m >>= 1) rs += __shfl_xor(rs, m, 32);
      lrow[e] = lrow[e] * alpha + rs;
      mrow[e] = mnew;
#pragma unroll
      for (int nt = 0; nt < 4; nt++) O[nt][e] *= alpha;
    }

    // D-layout -> A-layout re-fragmentation via per-wave LDS tile (DS in-order).
#pragma unroll
    for (int t = 0; t < 2; t++)
#pragma unroll
      for (int e = 0; e < 8; e++)
        Pw[(hi8 + e) * PSTR + t * 16 + nc] = f2bf(S[t][e]);
    const v16bf pf = load_a_frag(Pw, PSTR);

    // O += P(16x32) @ V(32xDh).  V is [key][dim] == [k][n] layout.
#pragma unroll
    for (int nt = 0; nt < 4; nt++) {
      v16bf vf = load_b_frag_kn(Vh + (size_t)j * DIM_HEAD + nt * 16, DIM_HEAD);
      O[nt] = __builtin_amdgcn_wmma_f32_16x16x32_bf16(
          false, pf, false, vf, (short)0, O[nt], false, false);
    }
  }

  // epilogue: normalize, emit bf16 [N][DINNER] for the out-projection GEMM
#pragma unroll
  for (int nt = 0; nt < 4; nt++)
#pragma unroll
    for (int e = 0; e < 8; e++) {
      const float o = O[nt][e] / lrow[e];
      ob[(size_t)(qrow0 + hi8 + e) * DINNER + h * DIM_HEAD + nt * 16 + nc] = f2bf(o);
    }
}

// ---------------------------------------------------------------------------
extern "C" void kernel_launch(void* const* d_in, const int* in_sizes, int n_in,
                              void* d_out, int out_size, void* d_ws, size_t ws_size,
                              hipStream_t stream) {
  const float* seq   = (const float*)d_in[0];
  const float* g     = (const float*)d_in[1];
  const float* w_qkv = (const float*)d_in[2];
  const float* w_out = (const float*)d_in[3];
  float* out = (float*)d_out;

  char* ws = (char*)d_ws;
  size_t off = 0;
  bf16* xb    = (bf16*)(ws + off); off += (size_t)N_SEQ * D_MODEL * 2;        // 8 MiB
  bf16* wqkvb = (bf16*)(ws + off); off += (size_t)D_MODEL * QKV_N * 2;        // 3 MiB
  bf16* woutb = (bf16*)(ws + off); off += (size_t)DINNER * D_MODEL * 2;       // 1 MiB
  float* qkvf = (float*)(ws + off); off += (size_t)N_SEQ * QKV_N * 4;         // 24 MiB
  bf16* qbuf  = (bf16*)(ws + off); off += (size_t)HEADS * N_SEQ * DIM_HEAD * 2;
  bf16* kbuf  = (bf16*)(ws + off); off += (size_t)HEADS * N_SEQ * DIM_HEAD * 2;
  bf16* vbuf  = (bf16*)(ws + off); off += (size_t)HEADS * N_SEQ * DIM_HEAD * 2;
  bf16* attnb = (bf16*)(ws + off); off += (size_t)N_SEQ * DINNER * 2;         // ~52 MiB total

  rmsnorm_kernel<<<N_SEQ, 256, 0, stream>>>(seq, g, xb);
  cvt_bf16_kernel<<<(D_MODEL * QKV_N + 255) / 256, 256, 0, stream>>>(w_qkv, wqkvb, D_MODEL * QKV_N);
  cvt_bf16_kernel<<<(DINNER * D_MODEL + 255) / 256, 256, 0, stream>>>(w_out, woutb, DINNER * D_MODEL);

  gemm_bf16_kernel<D_MODEL, QKV_N, QKV_N, D_MODEL>
      <<<dim3(QKV_N / BN, N_SEQ / BM), 256, 0, stream>>>(xb, wqkvb, qkvf);

  rope_split_kernel<<<N_SEQ, 256, 0, stream>>>(qkvf, qbuf, kbuf, vbuf);

  attn_kernel<<<dim3(N_SEQ / 64, HEADS), 128, 0, stream>>>(qbuf, kbuf, vbuf, attnb);

  gemm_bf16_kernel<DINNER, D_MODEL, D_MODEL, DINNER>
      <<<dim3(D_MODEL / BN, N_SEQ / BM), 256, 0, stream>>>(attnb, woutb, out);
}
